// BaselineNet_33200097198825
// MI455X (gfx1250) — compile-verified, matrix operand order checked by reading, not run on hardware
//
#include <hip/hip_runtime.h>
#include <hip/hip_bf16.h>
#include <math.h>

// ---------------- types for WMMA fragments (wave32, CDNA5) ----------------
typedef __attribute__((ext_vector_type(16))) _Float16 v16h;
typedef __attribute__((ext_vector_type(8)))  _Float16 v8h;
typedef __attribute__((ext_vector_type(8)))  float    v8f;

union V16 { v16h v; v8h h[2]; };

// ---------------- helpers ----------------
__device__ __forceinline__ float qz(float w, float s) {
  // symmetric 4-bit fake quant; rintf == round-half-even == jnp.round
  float q = rintf(w / s);
  q = fminf(fmaxf(q, -7.f), 7.f);
  return q * s;
}

__device__ __forceinline__ void atomicMaxF(float* addr, float val) {
  // ordered-float max via integer atomics (buffer initialized to -inf)
  if (val >= 0.f) atomicMax((int*)addr, __float_as_int(val));
  else            atomicMin((unsigned int*)addr, __float_as_uint(val));
}

// K index held by (lane, t) for 16-bit A/B fragments (ISA 7.12.2):
// lane<16: K = t<8 ? t : t+8 ; lane>=16: K = t<8 ? t+8 : t+16
__device__ __forceinline__ int fragK(int lane, int t) {
  return (lane < 16) ? (t < 8 ? t : t + 8) : (t < 8 ? t + 8 : t + 16);
}

// low 32 bits of a generic pointer to __shared__ == LDS byte offset
__device__ __forceinline__ unsigned ldsOffsetOf(const void* p) {
  return (unsigned)(uintptr_t)p;
}

// async global -> LDS copy of 16 bytes (gfx1250, ASYNCcnt-tracked)
__device__ __forceinline__ void asyncCopy16(unsigned ldsAddr, const void* gaddr) {
  unsigned long long ga = (unsigned long long)(uintptr_t)gaddr;
  asm volatile("global_load_async_to_lds_b128 %0, %1, off"
               :: "v"(ldsAddr), "v"(ga)
               : "memory");
}
__device__ __forceinline__ void waitAsync0() {
  asm volatile("s_wait_asynccnt 0x0" ::: "memory");
}

// ---------------- workspace layout (byte offsets) ----------------
#define WS_SCALES   0u            // 4 f32
#define WS_W1Q      256u          // 2400 f32
#define WS_WF2Q     16384u        // 512 f32
#define WS_W2FRAG   32768u        // 27648 f16   [27][2][32][16]
#define WS_WF1FRAG  131072u       // 884736 f16  [216][8][32][16]
#define WS_A1       (4ull<<20)    // 1024*128 f32
#define WS_F        (8ull<<20)    // 1024*6912 f32 (pooled features, NCDHW flat)
#define WS_C1       (40ull<<20)   // 1024*2744*32 f16 channels-last

// ---------------- quant scale reduction ----------------
__global__ void k_qscale(const float* __restrict__ w1, const float* __restrict__ w2,
                         const float* __restrict__ wf1, const float* __restrict__ wf2,
                         float* __restrict__ scales) {
  const float* p; int n;
  switch (blockIdx.x) {
    case 0:  p = w1;  n = 2400;   break;
    case 1:  p = w2;  n = 27648;  break;
    case 2:  p = wf1; n = 884736; break;
    default: p = wf2; n = 512;    break;
  }
  float m = 0.f;
  for (int i = threadIdx.x; i < n; i += blockDim.x) m = fmaxf(m, fabsf(p[i]));
  __shared__ float red[256];
  red[threadIdx.x] = m;
  __syncthreads();
  for (int s = 128; s > 0; s >>= 1) {
    if (threadIdx.x < (unsigned)s) red[threadIdx.x] = fmaxf(red[threadIdx.x], red[threadIdx.x + s]);
    __syncthreads();
  }
  if (threadIdx.x == 0) scales[blockIdx.x] = red[0] / 7.0f;
}

// ---------------- quantize / pack weights ----------------
__global__ void k_qpack_w1(const float* __restrict__ w1, const float* __restrict__ sc,
                           float* __restrict__ w1q) {
  int i = blockIdx.x * blockDim.x + threadIdx.x;
  if (i < 2400) w1q[i] = qz(w1[i], sc[0]);
}

// w2 (32o,32i,3,3,3) -> fragment layout [p][h][lane][t], B = (Kin x Nout) tile
__global__ void k_pack_w2(const float* __restrict__ w2, const float* __restrict__ sc,
                          _Float16* __restrict__ w2f) {
  int e = blockIdx.x * blockDim.x + threadIdx.x;
  if (e >= 27 * 2 * 32 * 16) return;
  int t = e & 15, L = (e >> 4) & 31, h = (e >> 9) & 1, p = e >> 10;
  int n = h * 16 + (L & 15);
  int kin = fragK(L, t);
  float w = w2[(n * 32 + kin) * 27 + p];
  w2f[e] = (_Float16)qz(w, sc[1]);
}

// wf1 (128,6912) -> fragment layout [ks][nt][lane][t]
__global__ void k_pack_wf1(const float* __restrict__ wf1, const float* __restrict__ sc,
                           _Float16* __restrict__ wf1f) {
  int e = blockIdx.x * blockDim.x + threadIdx.x;
  if (e >= 216 * 8 * 32 * 16) return;
  int t = e & 15, L = (e >> 4) & 31, nt = (e >> 9) & 7, ks = e >> 12;
  int n = nt * 16 + (L & 15);
  int k = ks * 32 + fragK(L, t);
  wf1f[e] = (_Float16)qz(wf1[n * 6912 + k], sc[2]);
}

__global__ void k_qpack_wf2(const float* __restrict__ wf2, const float* __restrict__ sc,
                            float* __restrict__ wf2q) {
  int i = blockIdx.x * blockDim.x + threadIdx.x;
  if (i < 512) wf2q[i] = qz(wf2[i], sc[3]);
}

__global__ void k_initf(unsigned int* __restrict__ f) {
  size_t i = (size_t)blockIdx.x * blockDim.x + threadIdx.x;
  f[i] = 0xFF800000u;  // -inf
}

// ---------------- cv1: direct conv, output channels-last f16 ----------------
__global__ void k_cv1(const float* __restrict__ x, const float* __restrict__ w1q,
                      const float* __restrict__ b1, _Float16* __restrict__ c1) {
  __shared__ float wl[2400];
  for (int i = threadIdx.x; i < 2400; i += blockDim.x) wl[i] = w1q[i];
  __syncthreads();
  int pix = blockIdx.x * blockDim.x + threadIdx.x;   // 1024*2744 exact
  int b = pix / 2744, r = pix % 2744;
  int z = r / 196, y = (r / 14) % 14, xo = r % 14;
  const float* xb = x + (size_t)b * 16384 + (2 * z) * 512 + y * 32 + 2 * xo;
  float acc[32];
#pragma unroll
  for (int c = 0; c < 32; ++c) acc[c] = 0.f;
  for (int kd = 0; kd < 5; ++kd)
    for (int kh = 0; kh < 3; ++kh)
#pragma unroll
      for (int kw = 0; kw < 5; ++kw) {
        float in = xb[kd * 512 + kh * 32 + kw];
        int tap = (kd * 3 + kh) * 5 + kw;
#pragma unroll
        for (int c = 0; c < 32; ++c) acc[c] = fmaf(in, wl[c * 75 + tap], acc[c]);
      }
  _Float16* o = c1 + (size_t)pix * 32;
#pragma unroll
  for (int c0 = 0; c0 < 32; c0 += 8) {
    v8h v;
#pragma unroll
    for (int j = 0; j < 8; ++j) v[j] = (_Float16)(acc[c0 + j] + b1[c0 + j]);
    *(v8h*)(o + c0) = v;
  }
}

// ---------------- cv2: async-LDS staged implicit-GEMM WMMA + fused maxpool ----------------
// Block tile: one batch b, output z in [z0, z0+2), full y (12), full x (12) = 288 pixels.
// Input needed: 4 contiguous z-slices of channels-last c1 = 4*14*14*32 f16 = 50 KB.
#define CV2_TILE_HALVES (4 * 14 * 14 * 32)   // 25088
__global__ void k_cv2(const _Float16* __restrict__ c1, const _Float16* __restrict__ w2f,
                      const float* __restrict__ b2, float* __restrict__ f) {
  __shared__ _Float16 tile[CV2_TILE_HALVES];

  int b  = blockIdx.x / 6;
  int z0 = (blockIdx.x % 6) * 2;

  // ---- async global -> LDS fill: 25088 halves = 3136 x 16B chunks ----
  const _Float16* gbase = c1 + ((size_t)b * 14 + z0) * 6272;   // 14*14*32 = 6272
  unsigned ldsBase = ldsOffsetOf(tile);
  for (int i = threadIdx.x; i < CV2_TILE_HALVES / 8; i += 256)
    asyncCopy16(ldsBase + i * 16, gbase + i * 8);
  waitAsync0();
  __syncthreads();

  int lane = threadIdx.x & 31;
  int wave = threadIdx.x >> 5;
  int koff = (lane < 16) ? 0 : 8;
  int nlo  = lane & 15;
  int m0   = (lane < 16) ? 0 : 8;
  float bias0 = b2[nlo], bias1 = b2[16 + nlo];

  // 288 local pixels -> 18 wave-tiles of 16; 8 waves iterate (wave-uniform guard)
  for (int it = 0; it < 3; ++it) {
    int tix = wave + it * 8;
    if (tix >= 18) break;                       // whole-wave, EXEC stays all-ones
    int q  = tix * 16 + nlo;                    // A-fragment row pixel
    int lz = q / 144, ly = (q / 12) % 12, lx = q % 12;
    const _Float16* aBase = tile + ((lz * 14 + ly) * 14 + lx) * 32 + koff;

    v8f acc0 = {}; v8f acc1 = {};
    for (int p = 0; p < 27; ++p) {
      int kd = p / 9, kh = (p / 3) % 3, kw = p % 3;
      const _Float16* ap = aBase + ((kd * 14 + kh) * 14 + kw) * 32;
      V16 A; A.h[0] = *(const v8h*)ap; A.h[1] = *(const v8h*)(ap + 16);
      const _Float16* bp = w2f + (size_t)(p * 2) * 512 + lane * 16;
      V16 B0, B1;
      B0.v = *(const v16h*)bp;
      B1.v = *(const v16h*)(bp + 512);
      acc0 = __builtin_amdgcn_wmma_f32_16x16x32_f16(false, A.v, false, B0.v, (short)0, acc0, false, false);
      acc1 = __builtin_amdgcn_wmma_f32_16x16x32_f16(false, A.v, false, B1.v, (short)0, acc1, false, false);
    }

    // D layout: lane<16 -> rows 0..7 col=lane ; lane>=16 -> rows 8..15 col=lane-16
#pragma unroll
    for (int rr = 0; rr < 8; ++rr) {
      int oq = tix * 16 + m0 + rr;
      int oz = z0 + oq / 144, oy = (oq / 12) % 12, ox = oq % 12;
      size_t fb = (size_t)b * 6912 + (oz >> 1) * 36 + (oy >> 1) * 6 + (ox >> 1);
      atomicMaxF(&f[fb + (size_t)nlo * 216], acc0[rr] + bias0);
      atomicMaxF(&f[fb + (size_t)(nlo + 16) * 216], acc1[rr] + bias1);
    }
  }
}

// ---------------- fc1: WMMA GEMM (1024x128, K=6912) + relu ----------------
__global__ void k_fc1(const float* __restrict__ f, const _Float16* __restrict__ wf1f,
                      const float* __restrict__ bf1, float* __restrict__ a1) {
  int lane = threadIdx.x & 31;
  int nt = threadIdx.x >> 5;                   // n-tile 0..7 -> cols nt*16..
  int mbase = blockIdx.x * 16;
  const float* arow = f + (size_t)(mbase + (lane & 15)) * 6912 + ((lane < 16) ? 0 : 8);
  v8f acc = {};
  for (int ks = 0; ks < 216; ++ks) {
    const float* ap = arow + ks * 32;
    V16 A;
#pragma unroll
    for (int j = 0; j < 8; ++j) A.h[0][j] = (_Float16)ap[j];
#pragma unroll
    for (int j = 0; j < 8; ++j) A.h[1][j] = (_Float16)ap[16 + j];
    V16 Bf; Bf.v = *(const v16h*)(wf1f + ((size_t)(ks * 8 + nt) * 32 + lane) * 16);
    acc = __builtin_amdgcn_wmma_f32_16x16x32_f16(false, A.v, false, Bf.v, (short)0, acc, false, false);
  }
  int n = nt * 16 + (lane & 15);
  int m0 = (lane < 16) ? 0 : 8;
  float bias = bf1[n];
#pragma unroll
  for (int rr = 0; rr < 8; ++rr) {
    float v = acc[rr] + bias;
    a1[(size_t)(mbase + m0 + rr) * 128 + n] = v > 0.f ? v : 0.f;
  }
}

// ---------------- fc2 + softmax ----------------
__global__ void k_fc2(const float* __restrict__ a1, const float* __restrict__ wf2q,
                      const float* __restrict__ bf2, float* __restrict__ out) {
  int row = blockIdx.x * blockDim.x + threadIdx.x;
  if (row >= 1024) return;
  const float* ar = a1 + (size_t)row * 128;
  float s0 = bf2[0], s1 = bf2[1], s2 = bf2[2], s3 = bf2[3];
  for (int k = 0; k < 128; ++k) {
    float a = ar[k];
    s0 = fmaf(a, wf2q[k], s0);
    s1 = fmaf(a, wf2q[128 + k], s1);
    s2 = fmaf(a, wf2q[256 + k], s2);
    s3 = fmaf(a, wf2q[384 + k], s3);
  }
  float m = fmaxf(fmaxf(s0, s1), fmaxf(s2, s3));
  float e0 = expf(s0 - m), e1 = expf(s1 - m), e2 = expf(s2 - m), e3 = expf(s3 - m);
  float inv = 1.f / (e0 + e1 + e2 + e3);
  out[row * 4 + 0] = e0 * inv;
  out[row * 4 + 1] = e1 * inv;
  out[row * 4 + 2] = e2 * inv;
  out[row * 4 + 3] = e3 * inv;
}

// ---------------- launcher ----------------
extern "C" void kernel_launch(void* const* d_in, const int* in_sizes, int n_in,
                              void* d_out, int out_size, void* d_ws, size_t ws_size,
                              hipStream_t stream) {
  (void)in_sizes; (void)n_in; (void)out_size; (void)ws_size;
  const float* x   = (const float*)d_in[0];
  const float* w1  = (const float*)d_in[1];
  const float* b1  = (const float*)d_in[2];
  const float* w2  = (const float*)d_in[3];
  const float* b2  = (const float*)d_in[4];
  const float* wf1 = (const float*)d_in[5];
  const float* bf1 = (const float*)d_in[6];
  const float* wf2 = (const float*)d_in[7];
  const float* bf2 = (const float*)d_in[8];
  float* out = (float*)d_out;

  char* ws = (char*)d_ws;
  float*    scales = (float*)(ws + WS_SCALES);
  float*    w1q    = (float*)(ws + WS_W1Q);
  float*    wf2q   = (float*)(ws + WS_WF2Q);
  _Float16* w2f    = (_Float16*)(ws + WS_W2FRAG);
  _Float16* wf1f   = (_Float16*)(ws + WS_WF1FRAG);
  float*    a1     = (float*)(ws + WS_A1);
  float*    f      = (float*)(ws + WS_F);
  _Float16* c1     = (_Float16*)(ws + WS_C1);

  k_qscale   <<<4,     256, 0, stream>>>(w1, w2, wf1, wf2, scales);
  k_qpack_w1 <<<10,    256, 0, stream>>>(w1, scales, w1q);
  k_pack_w2  <<<108,   256, 0, stream>>>(w2, scales, w2f);
  k_pack_wf1 <<<3456,  256, 0, stream>>>(wf1, scales, wf1f);
  k_qpack_wf2<<<2,     256, 0, stream>>>(wf2, scales, wf2q);
  k_initf    <<<27648, 256, 0, stream>>>((unsigned int*)f);       // 1024*6912 exact
  k_cv1      <<<10976, 256, 0, stream>>>(x, w1q, b1, c1);         // 1024*2744 exact
  k_cv2      <<<6144,  256, 0, stream>>>(c1, w2f, b2, f);         // 1024 b x 6 z-tiles
  k_fc1      <<<64,    256, 0, stream>>>(f, wf1f, bf1, a1);       // 1024/16 m-tiles
  k_fc2      <<<4,     256, 0, stream>>>(a1, wf2q, bf2, out);
}
